// Gemma4TextAttention_73882027425957
// MI455X (gfx1250) — matrix operand order repaired and checked by reading, not compiled, not run on hardware
//
#include <hip/hip_runtime.h>
#include <hip/hip_bf16.h>
#include <stdint.h>
#include <stddef.h>

#define B_    2
#define S_    2048
#define HID_  2560
#define HQ_   8
#define HKV_  4
#define HD_   256
#define WIN_  1024
#define NEGF  (-1000000000.0f)
#define EPSF  (1e-6f)

typedef __attribute__((ext_vector_type(16))) __bf16 v16bf;
typedef __attribute__((ext_vector_type(8)))  float  v8f;

struct U16x16 { unsigned short v[16]; };

__device__ __forceinline__ unsigned short f2bf(float f) {
  unsigned u = __builtin_bit_cast(unsigned, f);
  u += 0x7FFFu + ((u >> 16) & 1u);       // round-to-nearest-even to bf16
  return (unsigned short)(u >> 16);
}

// ---------------------------------------------------------------------------
// Pack hidden_states (f32 [B*S, HID]) into WMMA A-fragment order, bf16.
// Entry t = (rowtile*80 + ktile)*32 + lane; 16 bf16 per entry (32B aligned).
// A 16x32 bf16 layout: lane L -> row = L%16 ; k = (L/16)*16 + e
// ---------------------------------------------------------------------------
__global__ __launch_bounds__(256) void pack_a_hidden(const float* __restrict__ hs,
                                                     unsigned short* __restrict__ out) {
  const int NKT = HID_ / 32;             // 80
  const int total = (B_ * S_ / 16) * NKT * 32;
  int t = blockIdx.x * blockDim.x + threadIdx.x;
  if (t >= total) return;
  int lane = t & 31;
  int kt   = (t >> 5) % NKT;
  int rt   = t / (32 * NKT);
  int row  = rt * 16 + (lane & 15);
  int kb   = kt * 32 + (lane >> 4) * 16;
  U16x16 p;
#pragma unroll
  for (int e = 0; e < 16; ++e) {
    int k = kb + e;
    p.v[e] = f2bf(hs[(size_t)row * HID_ + k]);
  }
  *reinterpret_cast<v16bf*>(out + (size_t)t * 16) = __builtin_bit_cast(v16bf, p);
}

// ---------------------------------------------------------------------------
// Pack weight W (f32 [N, K] row-major; GEMM is X @ W^T) into WMMA B-fragment
// order, bf16.  B 32x16 layout: lane L -> col = L%16 ; k = (L/16)*16 + e
// Entry t = (ntile*(K/32) + ktile)*32 + lane.
// ---------------------------------------------------------------------------
__global__ __launch_bounds__(256) void pack_b_weight(const float* __restrict__ W,
                                                     unsigned short* __restrict__ out,
                                                     int N, int K) {
  const int NKT = K / 32;
  const int total = (N / 16) * NKT * 32;
  int t = blockIdx.x * blockDim.x + threadIdx.x;
  if (t >= total) return;
  int lane = t & 31;
  int nt   = t / (32 * NKT);
  int n    = nt * 16 + (lane & 15);
  int kb   = ((t >> 5) % NKT) * 32 + (lane >> 4) * 16;
  U16x16 p;
#pragma unroll
  for (int e = 0; e < 16; ++e) p.v[e] = f2bf(W[(size_t)n * K + kb + e]);
  *reinterpret_cast<v16bf*>(out + (size_t)t * 16) = __builtin_bit_cast(v16bf, p);
}

// ---------------------------------------------------------------------------
// Fused QKV projection + RMSNorm (+ RoPE for Q/K) + fragment-order store.
// grid = (B*S/16, 16): y<8 -> Q head y ; y<12 -> K head y-8 ; else V head y-12
// Block = 256 threads (8 waves); each wave owns 32 output columns; workgroup
// tile = 16 rows x 256 cols (= one head's full HD).
// ---------------------------------------------------------------------------
__global__ __launch_bounds__(256) void qkv_kernel(
    const unsigned short* __restrict__ hspack,
    const unsigned short* __restrict__ wq,
    const unsigned short* __restrict__ wk,
    const unsigned short* __restrict__ wv,
    const float* __restrict__ cosp, const float* __restrict__ sinp,
    const float* __restrict__ qnw,  const float* __restrict__ knw,
    unsigned short* __restrict__ qpack,
    unsigned short* __restrict__ kpack,
    unsigned short* __restrict__ vpack) {
  __shared__ float tile [16 * 256];
  __shared__ float tile2[16 * 256];
  __shared__ float part [16 * 16];
  __shared__ float rstd [16];

  const int rt    = blockIdx.x;
  const int which = blockIdx.y;
  const int tid   = threadIdx.x;
  const int wave  = tid >> 5;
  const int lane  = tid & 31;
  const int hf    = lane >> 4;
  const int n     = lane & 15;
  const int NKT   = HID_ / 32;           // 80

  const unsigned short* wp;
  int head, type;                         // 0=Q 1=K 2=V
  if (which < 8)       { type = 0; head = which;      wp = wq; }
  else if (which < 12) { type = 1; head = which - 8;  wp = wk; }
  else                 { type = 2; head = which - 12; wp = wv; }

  const int nt0 = head * 16 + wave * 2;

  v8f acc0, acc1;
#pragma unroll
  for (int r = 0; r < 8; ++r) { acc0[r] = 0.f; acc1[r] = 0.f; }

  const unsigned short* ap  = hspack + ((size_t)(rt * NKT) * 32 + lane) * 16;
  const unsigned short* bp0 = wp + ((size_t)(nt0       * NKT) * 32 + lane) * 16;
  const unsigned short* bp1 = wp + ((size_t)((nt0 + 1) * NKT) * 32 + lane) * 16;

  for (int kt = 0; kt < NKT; ++kt) {
    v16bf a  = *reinterpret_cast<const v16bf*>(ap  + (size_t)kt * 512);
    v16bf b0 = *reinterpret_cast<const v16bf*>(bp0 + (size_t)kt * 512);
    v16bf b1 = *reinterpret_cast<const v16bf*>(bp1 + (size_t)kt * 512);
    acc0 = __builtin_amdgcn_wmma_f32_16x16x32_bf16(false, a, false, b0, (short)0, acc0, false, false);
    acc1 = __builtin_amdgcn_wmma_f32_16x16x32_bf16(false, a, false, b1, (short)0, acc1, false, false);
  }

  // C-fragment (lane -> row r + hf*8, col n) into LDS tile
  const int c0 = wave * 32;
#pragma unroll
  for (int r = 0; r < 8; ++r) {
    int row = r + hf * 8;
    tile[row * 256 + c0 + n]      = acc0[r];
    tile[row * 256 + c0 + 16 + n] = acc1[r];
  }
  __syncthreads();

  // RMS statistics over the 256-wide head row
  {
    int row = tid >> 4, seg = tid & 15;
    float s = 0.f;
#pragma unroll
    for (int c = 0; c < 16; ++c) { float x = tile[row * 256 + seg * 16 + c]; s += x * x; }
    part[row * 16 + seg] = s;
  }
  __syncthreads();
  if (tid < 16) {
    float s = 0.f;
#pragma unroll
    for (int g = 0; g < 16; ++g) s += part[tid * 16 + g];
    rstd[tid] = rsqrtf(s * (1.f / 256.f) + EPSF);
  }
  __syncthreads();

  // Apply norm (+ RoPE for Q/K)
  {
    int row = tid >> 4, seg = tid & 15;
    int grow = rt * 16 + row;
    float rs = rstd[row];
    if (type == 2) {
#pragma unroll
      for (int c = 0; c < 16; ++c) {
        int d = seg * 16 + c;
        tile2[row * 256 + d] = tile[row * 256 + d] * rs;
      }
    } else {
      const float* w = (type == 0) ? qnw : knw;
#pragma unroll
      for (int c = 0; c < 16; ++c) {
        int d  = seg * 16 + c;
        int dp = (d + 128) & 255;
        float x  = tile[row * 256 + d]  * rs * w[d];
        float xp = tile[row * 256 + dp] * rs * w[dp];
        float cv = cosp[(size_t)grow * HD_ + d];
        float sv = sinp[(size_t)grow * HD_ + d];
        tile2[row * 256 + d] = (d < 128) ? (x * cv - xp * sv) : (x * cv + xp * sv);
      }
    }
  }
  __syncthreads();

  const int b  = rt / (S_ / 16);
  const int pt = rt % (S_ / 16);

  if (type == 0) {
    // Q in A-fragment order: [bh][pt][dt32 0..7][lane][e]
    int dt = tid >> 5;
    U16x16 p;
#pragma unroll
    for (int e = 0; e < 16; ++e) {
      int k = dt * 32 + hf * 16 + e;
      p.v[e] = f2bf(tile2[(lane & 15) * 256 + k]);
    }
    size_t base = ((((size_t)(b * HQ_ + head) * (S_ / 16) + pt) * 8 + dt) * 32 + lane) * 16;
    *reinterpret_cast<v16bf*>(qpack + base) = __builtin_bit_cast(v16bf, p);
  } else if (type == 1) {
    // K^T in B-fragment order: [bhk][pt][dt32 0..7][lane][e]; col=key, k=d
    int dt = tid >> 5;
    U16x16 p;
#pragma unroll
    for (int e = 0; e < 16; ++e)
      p.v[e] = f2bf(tile2[(lane & 15) * 256 + dt * 32 + hf * 16 + e]);
    size_t base = ((((size_t)(b * HKV_ + head) * (S_ / 16) + pt) * 8 + dt) * 32 + lane) * 16;
    *reinterpret_cast<v16bf*>(kpack + base) = __builtin_bit_cast(v16bf, p);
  } else {
    // V in B-fragment half-order: [bhk][pt][dt16 0..15][n][e]; col=d, k=key
    int dt = tid >> 4, nn = tid & 15;
    U16x16 p;
#pragma unroll
    for (int e = 0; e < 16; ++e)
      p.v[e] = f2bf(tile2[e * 256 + dt * 16 + nn]);
    size_t base = ((((size_t)(b * HKV_ + head) * (S_ / 16) + pt) * 16 + dt) * 256 + nn * 16);
    *reinterpret_cast<v16bf*>(vpack + base) = __builtin_bit_cast(v16bf, p);
  }
}

// ---------------------------------------------------------------------------
// Flash attention, sliding window. One wave (32 threads) per 16-query tile.
// grid = (S/16, B*HQ). Mask computed analytically (skip reading 67MB mask).
// ---------------------------------------------------------------------------
__global__ __launch_bounds__(32) void attn_kernel(
    const unsigned short* __restrict__ qpack,
    const unsigned short* __restrict__ kpack,
    const unsigned short* __restrict__ vpack,
    float* __restrict__ attnbuf) {
  __shared__ float plds[16 * 32];
  const int ptq  = blockIdx.x;
  const int bh   = blockIdx.y;
  const int b    = bh / HQ_, h = bh % HQ_;
  const int bhk  = b * HKV_ + (h >> 1);          // n_rep = 2
  const int q0   = ptq * 16;
  const int lane = threadIdx.x;
  const int hf   = lane >> 4, n = lane & 15;
  const float scale = 0.0625f;                    // HD^-0.5

  v16bf qf[8];
#pragma unroll
  for (int dt = 0; dt < 8; ++dt)
    qf[dt] = *reinterpret_cast<const v16bf*>(
        qpack + ((((size_t)bh * (S_ / 16) + ptq) * 8 + dt) * 32 + lane) * 16);

  v8f acc[16];
#pragma unroll
  for (int dt = 0; dt < 16; ++dt)
#pragma unroll
    for (int r = 0; r < 8; ++r) acc[dt][r] = 0.f;

  float mi[8], li[8];
#pragma unroll
  for (int r = 0; r < 8; ++r) { mi[r] = -3e38f; li[r] = 0.f; }

  int lo = q0 - WIN_ + 1; if (lo < 0) lo = 0;
  const int kt0 = lo & ~31;

  for (int kt = kt0; kt < q0 + 16; kt += 32) {
    const int ptk = kt >> 4;
    v8f s0, s1;
#pragma unroll
    for (int r = 0; r < 8; ++r) { s0[r] = 0.f; s1[r] = 0.f; }
#pragma unroll
    for (int dt = 0; dt < 8; ++dt) {
      v16bf k0 = *reinterpret_cast<const v16bf*>(
          kpack + ((((size_t)bhk * (S_ / 16) + ptk)     * 8 + dt) * 32 + lane) * 16);
      v16bf k1 = *reinterpret_cast<const v16bf*>(
          kpack + ((((size_t)bhk * (S_ / 16) + ptk + 1) * 8 + dt) * 32 + lane) * 16);
      s0 = __builtin_amdgcn_wmma_f32_16x16x32_bf16(false, qf[dt], false, k0, (short)0, s0, false, false);
      s1 = __builtin_amdgcn_wmma_f32_16x16x32_bf16(false, qf[dt], false, k1, (short)0, s1, false, false);
    }
    // analytic sliding-window mask + online softmax (per-row, shuffle reduce
    // within each 16-lane half which holds the 16 columns of each row)
    float alpha[8];
#pragma unroll
    for (int r = 0; r < 8; ++r) {
      int i  = q0 + r + hf * 8;
      int j0 = kt + n, j1 = kt + 16 + n;
      s0[r] = ((j0 <= i) && (j0 > i - WIN_)) ? s0[r] * scale : NEGF;
      s1[r] = ((j1 <= i) && (j1 > i - WIN_)) ? s1[r] * scale : NEGF;
      float rm = fmaxf(s0[r], s1[r]);
      rm = fmaxf(rm, __shfl_xor(rm, 1, 32));
      rm = fmaxf(rm, __shfl_xor(rm, 2, 32));
      rm = fmaxf(rm, __shfl_xor(rm, 4, 32));
      rm = fmaxf(rm, __shfl_xor(rm, 8, 32));
      float mn = fmaxf(mi[r], rm);
      float al = __expf(mi[r] - mn);
      float p0 = (s0[r] > -1e8f) ? __expf(s0[r] - mn) : 0.f;
      float p1 = (s1[r] > -1e8f) ? __expf(s1[r] - mn) : 0.f;
      float rs = p0 + p1;
      rs += __shfl_xor(rs, 1, 32);
      rs += __shfl_xor(rs, 2, 32);
      rs += __shfl_xor(rs, 4, 32);
      rs += __shfl_xor(rs, 8, 32);
      li[r] = li[r] * al + rs;
      mi[r] = mn; alpha[r] = al;
      s0[r] = p0; s1[r] = p1;
    }
#pragma unroll
    for (int dt = 0; dt < 16; ++dt)
#pragma unroll
      for (int r = 0; r < 8; ++r) acc[dt][r] *= alpha[r];

    // P (two C-fragments) -> one 16x32 A-fragment via LDS relayout
    __syncthreads();
#pragma unroll
    for (int r = 0; r < 8; ++r) {
      plds[(r + hf * 8) * 32 + n]      = s0[r];
      plds[(r + hf * 8) * 32 + 16 + n] = s1[r];
    }
    __syncthreads();
    U16x16 t;
#pragma unroll
    for (int e = 0; e < 16; ++e)
      t.v[e] = f2bf(plds[(lane & 15) * 32 + hf * 16 + e]);
    v16bf pa = __builtin_bit_cast(v16bf, t);

#pragma unroll
    for (int dt = 0; dt < 16; ++dt) {
      v16bf vb = *reinterpret_cast<const v16bf*>(
          vpack + ((((size_t)bhk * (S_ / 16) + ptk + hf) * 16 + dt) * 256 + n * 16));
      acc[dt] = __builtin_amdgcn_wmma_f32_16x16x32_bf16(false, pa, false, vb, (short)0, acc[dt], false, false);
    }
  }

  // normalize and store f32 [B*S, HQ*HD]
#pragma unroll
  for (int r = 0; r < 8; ++r) {
    float inv = 1.f / li[r];
    size_t grow = (size_t)b * S_ + q0 + r + hf * 8;
#pragma unroll
    for (int dt = 0; dt < 16; ++dt)
      attnbuf[grow * (HQ_ * HD_) + h * HD_ + dt * 16 + n] = acc[dt][r] * inv;
  }
}

// ---------------------------------------------------------------------------
// Output projection: out[B*S, HID] = attn[B*S, 2048] @ Wo^T (packed bf16 B).
// grid = (B*S/16, HID/256); block 256 (8 waves x 32 cols each).
// ---------------------------------------------------------------------------
__global__ __launch_bounds__(256) void out_gemm(
    const float* __restrict__ attnbuf,
    const unsigned short* __restrict__ wopack,
    float* __restrict__ out) {
  const int rt   = blockIdx.x;
  const int cg   = blockIdx.y;
  const int tid  = threadIdx.x;
  const int wave = tid >> 5, lane = tid & 31;
  const int hf   = lane >> 4, n = lane & 15;
  const int NKT  = (HQ_ * HD_) / 32;     // 64
  const int nt0  = cg * 16 + wave * 2;

  v8f acc0, acc1;
#pragma unroll
  for (int r = 0; r < 8; ++r) { acc0[r] = 0.f; acc1[r] = 0.f; }

  const float* arow = attnbuf + (size_t)(rt * 16 + (lane & 15)) * (HQ_ * HD_);

  for (int kt = 0; kt < NKT; ++kt) {
    int kb = kt * 32 + hf * 16;
    float4 f0 = *reinterpret_cast<const float4*>(arow + kb);
    float4 f1 = *reinterpret_cast<const float4*>(arow + kb + 4);
    float4 f2 = *reinterpret_cast<const float4*>(arow + kb + 8);
    float4 f3 = *reinterpret_cast<const float4*>(arow + kb + 12);
    U16x16 t;
    t.v[0]=f2bf(f0.x);  t.v[1]=f2bf(f0.y);  t.v[2]=f2bf(f0.z);  t.v[3]=f2bf(f0.w);
    t.v[4]=f2bf(f1.x);  t.v[5]=f2bf(f1.y);  t.v[6]=f2bf(f1.z);  t.v[7]=f2bf(f1.w);
    t.v[8]=f2bf(f2.x);  t.v[9]=f2bf(f2.y);  t.v[10]=f2bf(f2.z); t.v[11]=f2bf(f2.w);
    t.v[12]=f2bf(f3.x); t.v[13]=f2bf(f3.y); t.v[14]=f2bf(f3.z); t.v[15]=f2bf(f3.w);
    v16bf a  = __builtin_bit_cast(v16bf, t);
    v16bf b0 = *reinterpret_cast<const v16bf*>(wopack + (((size_t)nt0       * NKT + kt) * 32 + lane) * 16);
    v16bf b1 = *reinterpret_cast<const v16bf*>(wopack + (((size_t)(nt0 + 1) * NKT + kt) * 32 + lane) * 16);
    acc0 = __builtin_amdgcn_wmma_f32_16x16x32_bf16(false, a, false, b0, (short)0, acc0, false, false);
    acc1 = __builtin_amdgcn_wmma_f32_16x16x32_bf16(false, a, false, b1, (short)0, acc1, false, false);
  }
#pragma unroll
  for (int r = 0; r < 8; ++r) {
    size_t orow = (size_t)rt * 16 + r + hf * 8;
    out[orow * HID_ + nt0 * 16 + n]        = acc0[r];
    out[orow * HID_ + (nt0 + 1) * 16 + n]  = acc1[r];
  }
}

// ---------------------------------------------------------------------------
extern "C" void kernel_launch(void* const* d_in, const int* in_sizes, int n_in,
                              void* d_out, int out_size, void* d_ws, size_t ws_size,
                              hipStream_t stream) {
  (void)in_sizes; (void)n_in; (void)out_size; (void)ws_size;
  const float* hs  = (const float*)d_in[0];
  const float* cs  = (const float*)d_in[1];
  const float* sn  = (const float*)d_in[2];
  // d_in[3] attention_mask: reconstructed analytically, not read
  const float* Wq  = (const float*)d_in[4];
  const float* Wk  = (const float*)d_in[5];
  const float* Wv  = (const float*)d_in[6];
  const float* Wo  = (const float*)d_in[7];
  const float* qnw = (const float*)d_in[8];
  const float* knw = (const float*)d_in[9];
  float* out = (float*)d_out;

  char* ws = (char*)d_ws;
  size_t off = 0;
  unsigned short* hspack = (unsigned short*)(ws + off); off += (size_t)B_*S_*HID_*2;        // 21.0 MB
  unsigned short* wqp    = (unsigned short*)(ws + off); off += (size_t)HQ_*HD_*HID_*2;      // 10.5 MB
  unsigned short* wkp    = (unsigned short*)(ws + off); off += (size_t)HKV_*HD_*HID_*2;     //  5.2 MB
  unsigned short* wvp    = (unsigned short*)(ws + off); off += (size_t)HKV_*HD_*HID_*2;     //  5.2 MB
  unsigned short* wop    = (unsigned short*)(ws + off); off += (size_t)HID_*HQ_*HD_*2;      // 10.5 MB
  unsigned short* qp     = (unsigned short*)(ws + off); off += (size_t)B_*HQ_*S_*HD_*2;     // 16.8 MB
  unsigned short* kp     = (unsigned short*)(ws + off); off += (size_t)B_*HKV_*S_*HD_*2;    //  8.4 MB
  unsigned short* vp     = (unsigned short*)(ws + off); off += (size_t)B_*HKV_*S_*HD_*2;    //  8.4 MB
  float*          attnb  = (float*)(ws + off);          off += (size_t)B_*S_*HQ_*HD_*4;     // 33.6 MB

  // 1) pack activations and weights into WMMA fragment order (bf16)
  {
    int total = (B_*S_/16) * (HID_/32) * 32;
    pack_a_hidden<<<(total + 255)/256, 256, 0, stream>>>(hs, hspack);
  }
  {
    int t;
    t = (HQ_*HD_/16)  * (HID_/32) * 32;    pack_b_weight<<<(t+255)/256, 256, 0, stream>>>(Wq, wqp, HQ_*HD_,  HID_);
    t = (HKV_*HD_/16) * (HID_/32) * 32;    pack_b_weight<<<(t+255)/256, 256, 0, stream>>>(Wk, wkp, HKV_*HD_, HID_);
    t = (HKV_*HD_/16) * (HID_/32) * 32;    pack_b_weight<<<(t+255)/256, 256, 0, stream>>>(Wv, wvp, HKV_*HD_, HID_);
    t = (HID_/16) * ((HQ_*HD_)/32) * 32;   pack_b_weight<<<(t+255)/256, 256, 0, stream>>>(Wo, wop, HID_, HQ_*HD_);
  }
  // 2) fused QKV GEMM + RMSNorm + RoPE -> fragment-packed Q/K/V
  qkv_kernel<<<dim3(B_*S_/16, 16), 256, 0, stream>>>(hspack, wqp, wkp, wvp,
                                                     cs, sn, qnw, knw, qp, kp, vp);
  // 3) sliding-window flash attention (WMMA QK^T and PV)
  attn_kernel<<<dim3(S_/16, B_*HQ_), 32, 0, stream>>>(qp, kp, vp, attnb);
  // 4) output projection GEMM
  out_gemm<<<dim3(B_*S_/16, HID_/256), 256, 0, stream>>>(attnb, wop, out);
}
